// GCNLayer_309237645656
// MI455X (gfx1250) — matrix operand order repaired and checked by reading, not compiled, not run on hardware
//
#include <hip/hip_runtime.h>
#include <hip/hip_bf16.h>

typedef __attribute__((ext_vector_type(16))) __bf16 v16bf;
typedef __attribute__((ext_vector_type(8)))  float  v8f;

#define D 64
#define BN_EPS 1e-5f

union V16U { v16bf v; __bf16 b[16]; unsigned short u[16]; uint4 q[2]; };

__device__ __forceinline__ unsigned short f2bf(float f) {
    union { __bf16 b; unsigned short u; } cv;
    cv.b = (__bf16)f;                         // native cvt (RNE) on gfx1250
    return cv.u;
}
// monotone key: float -> uint preserving order under unsigned compare
__device__ __forceinline__ unsigned fkey(float f) {
    unsigned u = __float_as_uint(f);
    return ((int)u >= 0) ? (u | 0x80000000u) : ~u;
}
__device__ __forceinline__ float funkey(unsigned k) {
    unsigned u = (k & 0x80000000u) ? (k ^ 0x80000000u) : ~k;
    return __uint_as_float(u);
}

// ---- pack W[64][64] (row-major, row=K) into per-lane WMMA B fragments (bf16)
// frag index: ((kt*4 + nt)*32 + lane)*16 + i ; lane holds col = nt*16 + lane%16,
// K = kt*32 + (lane/16)*16 + i
__global__ void prep_bfrag(const float* __restrict__ Wc, const float* __restrict__ Wr,
                           unsigned short* __restrict__ bfC, unsigned short* __restrict__ bfR) {
    int t = threadIdx.x;                      // 256 threads = 2 kt * 4 nt * 32 lanes
    int kt = t >> 7, nt = (t >> 5) & 3, lane = t & 31;
    int col = nt * 16 + (lane & 15);
    int kbase = kt * 32 + ((lane >> 4) & 1) * 16;
    #pragma unroll
    for (int i = 0; i < 16; ++i) {
        int k = kbase + i;
        bfC[t * 16 + i] = f2bf(Wc[k * D + col]);
        bfR[t * 16 + i] = f2bf(Wr[k * D + col]);
    }
}

__global__ void zero_kernel(unsigned* __restrict__ smaxkey, float* __restrict__ denom,
                            float* __restrict__ stats, int nodes) {
    int i = blockIdx.x * blockDim.x + threadIdx.x;
    if (i < nodes) { smaxkey[i] = 0u; denom[i] = 0.f; }
    if (i < 128) stats[i] = 0.f;
}

// ---- ft = feat@W_conv + b_conv (-> ws), out = relu(feat@W_res + b_res) (-> d_out)
// one wave per 16-row tile, 4 N-tiles, K=64 as 2 x wmma 16x16x32 bf16 per matrix
__global__ void gemm_wmma(const float* __restrict__ feat,
                          const float* __restrict__ bconv, const float* __restrict__ bres,
                          const unsigned short* __restrict__ bfC,
                          const unsigned short* __restrict__ bfR,
                          float* __restrict__ ft, float* __restrict__ out, int nodes) {
    int wave = threadIdx.x >> 5;
    int lane = threadIdx.x & 31;
    int mtile = blockIdx.x * (blockDim.x >> 5) + wave;
    int mtiles = (nodes + 15) >> 4;
    if (mtile >= mtiles) return;              // whole-wave exit: EXEC stays all-1s for WMMA
    int m0 = mtile * 16;
    int half = (lane >> 4) & 1;
    int mrow = m0 + (lane & 15);
    if (mrow >= nodes) mrow = nodes - 1;      // clamp (select, no EXEC change)
    const float* fr = feat + (size_t)mrow * D;
    bool fulltile = (m0 + 16) <= nodes;       // wave-uniform

    // A fragments: 16-bit A 16x32 layout — elem i: K=16*half+i (contiguous, matches B frags)
    V16U A0, A1;
    #pragma unroll
    for (int i = 0; i < 8; ++i) {
        A0.b[i]     = (__bf16)fr[half * 16 + i];
        A0.b[8 + i] = (__bf16)fr[half * 16 + 8 + i];
        A1.b[i]     = (__bf16)fr[32 + half * 16 + i];
        A1.b[8 + i] = (__bf16)fr[32 + half * 16 + 8 + i];
    }

    #pragma unroll
    for (int nt = 0; nt < 4; ++nt) {
        int col = nt * 16 + (lane & 15);
        float bc = bconv[col], br = bres[col];
        v8f accC, accR;
        #pragma unroll
        for (int r = 0; r < 8; ++r) { accC[r] = bc; accR[r] = br; }

        V16U B0, B1;
        const uint4* p0 = reinterpret_cast<const uint4*>(bfC + ((size_t)(0 * 4 + nt) * 32 + lane) * 16);
        const uint4* p1 = reinterpret_cast<const uint4*>(bfC + ((size_t)(1 * 4 + nt) * 32 + lane) * 16);
        B0.q[0] = p0[0]; B0.q[1] = p0[1];
        B1.q[0] = p1[0]; B1.q[1] = p1[1];
        accC = __builtin_amdgcn_wmma_f32_16x16x32_bf16(false, A0.v, false, B0.v, (short)0, accC, false, false);
        accC = __builtin_amdgcn_wmma_f32_16x16x32_bf16(false, A1.v, false, B1.v, (short)0, accC, false, false);

        p0 = reinterpret_cast<const uint4*>(bfR + ((size_t)(0 * 4 + nt) * 32 + lane) * 16);
        p1 = reinterpret_cast<const uint4*>(bfR + ((size_t)(1 * 4 + nt) * 32 + lane) * 16);
        B0.q[0] = p0[0]; B0.q[1] = p0[1];
        B1.q[0] = p1[0]; B1.q[1] = p1[1];
        accR = __builtin_amdgcn_wmma_f32_16x16x32_bf16(false, A0.v, false, B0.v, (short)0, accR, false, false);
        accR = __builtin_amdgcn_wmma_f32_16x16x32_bf16(false, A1.v, false, B1.v, (short)0, accR, false, false);

        // C/D layout: VGPR r holds row m0 + 8*half + r, col = n0 + lane%16
        if (fulltile) {                       // uniform branch: common unguarded path
            #pragma unroll
            for (int r = 0; r < 8; ++r) {
                size_t row = (size_t)(m0 + half * 8 + r);
                ft[row * D + col]  = accC[r];
                out[row * D + col] = fmaxf(accR[r], 0.f);   // residual relu
            }
        } else {
            #pragma unroll
            for (int r = 0; r < 8; ++r) {
                int row = m0 + half * 8 + r;
                if (row < nodes) {
                    ft[(size_t)row * D + col]  = accC[r];
                    out[(size_t)row * D + col] = fmaxf(accR[r], 0.f);
                }
            }
        }
    }
}

// ---- per-edge dot score + segment max (8 lanes per edge)
__global__ void edge_score(const float* __restrict__ feat, const int* __restrict__ src,
                           const int* __restrict__ dst, float* __restrict__ score,
                           unsigned* __restrict__ smaxkey, int nedges) {
    int t = blockIdx.x * blockDim.x + threadIdx.x;
    int e = t >> 3, sub = t & 7;
    if (e >= nedges) return;
    int s = src[e], d = dst[e];
    const float4* ps = reinterpret_cast<const float4*>(feat + (size_t)s * D) + sub * 2;
    const float4* pd = reinterpret_cast<const float4*>(feat + (size_t)d * D) + sub * 2;
    float4 a0 = ps[0], a1 = ps[1], b0 = pd[0], b1 = pd[1];
    float p = a0.x * b0.x + a0.y * b0.y + a0.z * b0.z + a0.w * b0.w
            + a1.x * b1.x + a1.y * b1.y + a1.z * b1.z + a1.w * b1.w;
    p += __shfl_xor(p, 1, 8);
    p += __shfl_xor(p, 2, 8);
    p += __shfl_xor(p, 4, 8);
    if (sub == 0) {
        score[e] = p;
        atomicMax(smaxkey + d, fkey(p));
    }
}

// ---- ez = exp(score - smax[dst]); denom += ez
__global__ void edge_exp(const int* __restrict__ dst, const unsigned* __restrict__ smaxkey,
                         float* __restrict__ score, float* __restrict__ denom, int nedges) {
    int e = blockIdx.x * blockDim.x + threadIdx.x;
    if (e >= nedges) return;
    int d = dst[e];
    float ez = __expf(score[e] - funkey(smaxkey[d]));
    score[e] = ez;                            // overwrite in place
    atomicAdd(denom + d, ez);
}

// ---- out[dst] += ft[src] * (ez/denom[dst])  (16 lanes per edge, float4 each)
__global__ void edge_agg(const int* __restrict__ src, const int* __restrict__ dst,
                         const float* __restrict__ ez, const float* __restrict__ denom,
                         const float* __restrict__ ft, float* __restrict__ out, int nedges) {
    int t = blockIdx.x * blockDim.x + threadIdx.x;
    int e = t >> 4, sub = t & 15;
    if (e >= nedges) return;
    int s = src[e], d = dst[e];
    float a = ez[e] / denom[d];
    float4 v = reinterpret_cast<const float4*>(ft + (size_t)s * D)[sub];
    float* o = out + (size_t)d * D + sub * 4;
    atomicAdd(o + 0, v.x * a);
    atomicAdd(o + 1, v.y * a);
    atomicAdd(o + 2, v.z * a);
    atomicAdd(o + 3, v.w * a);
}

// ---- BN stats: per-channel sum & sumsq over out
__global__ void bn_stats(const float* __restrict__ out, float* __restrict__ stats, int nodes) {
    __shared__ float s_sum[256], s_sq[256];
    int c = threadIdx.x & 63, grp = threadIdx.x >> 6;
    float s = 0.f, s2 = 0.f;
    for (int row = blockIdx.x * 4 + grp; row < nodes; row += gridDim.x * 4) {
        float v = out[(size_t)row * D + c];
        s += v; s2 += v * v;
    }
    s_sum[threadIdx.x] = s; s_sq[threadIdx.x] = s2;
    __syncthreads();
    if (threadIdx.x < 64) {
        float ts = s_sum[threadIdx.x] + s_sum[threadIdx.x + 64] + s_sum[threadIdx.x + 128] + s_sum[threadIdx.x + 192];
        float t2 = s_sq[threadIdx.x]  + s_sq[threadIdx.x + 64]  + s_sq[threadIdx.x + 128]  + s_sq[threadIdx.x + 192];
        atomicAdd(stats + c, ts);
        atomicAdd(stats + 64 + c, t2);
    }
}

// ---- BN apply in place
__global__ void bn_apply(float* __restrict__ out, const float* __restrict__ stats,
                         const float* __restrict__ gamma, const float* __restrict__ beta,
                         int nodes) {
    int i = blockIdx.x * blockDim.x + threadIdx.x;
    if (i >= nodes * D) return;
    int c = i & 63;
    float inv_n = 1.f / (float)nodes;
    float mean = stats[c] * inv_n;
    float var  = stats[64 + c] * inv_n - mean * mean;
    float scale = rsqrtf(var + BN_EPS) * gamma[c];
    out[i] = (out[i] - mean) * scale + beta[c];
}

extern "C" void kernel_launch(void* const* d_in, const int* in_sizes, int n_in,
                              void* d_out, int out_size, void* d_ws, size_t ws_size,
                              hipStream_t stream) {
    const float* feat  = (const float*)d_in[0];
    const float* Wc    = (const float*)d_in[1];
    const float* bc    = (const float*)d_in[2];
    const float* Wr    = (const float*)d_in[3];
    const float* br    = (const float*)d_in[4];
    const float* gamma = (const float*)d_in[5];
    const float* beta  = (const float*)d_in[6];
    const int*   src   = (const int*)d_in[7];
    const int*   dst   = (const int*)d_in[8];
    int nodes  = in_sizes[0] / D;
    int nedges = in_sizes[7];
    float* out = (float*)d_out;

    // workspace carve-up (~33 MB)
    char* ws = (char*)d_ws;
    size_t off = 0;
    float* ft = (float*)(ws + off);            off += (size_t)nodes * D * sizeof(float);
    float* score = (float*)(ws + off);         off += (size_t)nedges * sizeof(float);
    unsigned* smaxkey = (unsigned*)(ws + off); off += (size_t)nodes * sizeof(unsigned);
    float* denom = (float*)(ws + off);         off += (size_t)nodes * sizeof(float);
    off = (off + 31) & ~(size_t)31;
    unsigned short* bfC = (unsigned short*)(ws + off); off += (size_t)2 * 4 * 32 * 16 * sizeof(unsigned short);
    unsigned short* bfR = (unsigned short*)(ws + off); off += (size_t)2 * 4 * 32 * 16 * sizeof(unsigned short);
    float* stats = (float*)(ws + off);         off += 128 * sizeof(float);

    hipLaunchKernelGGL(prep_bfrag, dim3(1), dim3(256), 0, stream, Wc, Wr, bfC, bfR);
    hipLaunchKernelGGL(zero_kernel, dim3((nodes + 255) / 256), dim3(256), 0, stream,
                       smaxkey, denom, stats, nodes);
    int mtiles = (nodes + 15) / 16;
    hipLaunchKernelGGL(gemm_wmma, dim3((mtiles + 7) / 8), dim3(256), 0, stream,
                       feat, bc, br, bfC, bfR, ft, out, nodes);
    hipLaunchKernelGGL(edge_score, dim3((unsigned)(((size_t)nedges * 8 + 255) / 256)), dim3(256), 0, stream,
                       feat, src, dst, score, smaxkey, nedges);
    hipLaunchKernelGGL(edge_exp, dim3((nedges + 255) / 256), dim3(256), 0, stream,
                       dst, smaxkey, score, denom, nedges);
    hipLaunchKernelGGL(edge_agg, dim3((unsigned)(((size_t)nedges * 16 + 255) / 256)), dim3(256), 0, stream,
                       src, dst, score, denom, ft, out, nedges);
    hipLaunchKernelGGL(bn_stats, dim3(512), dim3(256), 0, stream, out, stats, nodes);
    hipLaunchKernelGGL(bn_apply, dim3((nodes * D + 255) / 256), dim3(256), 0, stream,
                       out, stats, gamma, beta, nodes);
}
